// SCDMNSliced_6665789243880
// MI455X (gfx1250) — compile-verified, matrix operand order checked by reading, not run on hardware
//
#include <hip/hip_runtime.h>
#include <hip/hip_bf16.h>

// ---------------------------------------------------------------------------
// CDNA5 / gfx1250 sliced-ResNet. bf16 NHWC activations, gathered bf16 weights
// in [Cout][ky][kx][Cin] order so every 32-wide K chunk of the
// v_wmma_f32_16x16x32_bf16 implicit GEMM is a single conv tap with
// contiguous-in-Cin operands -> pure b128 loads, no conversion and no div/mod
// in the hot loop. Each wave32 computes a 32(Cout) x 32(pixel) tile with four
// f32 accumulators. Both K loops are kept rolled (#pragma unroll 1) so the
// live set (~80 VGPRs) never spills; latency is hidden by the thousands of
// independent single-wave workgroups per conv. The epilogue fuses BN /
// residual-add / ReLU, converting f32->bf16 once per element with 16-byte
// vector stores.
// ---------------------------------------------------------------------------

typedef __bf16 bf16;
typedef __attribute__((ext_vector_type(16))) __bf16 v16bf;
typedef __attribute__((ext_vector_type(8)))  __bf16 v8bf;
typedef __attribute__((ext_vector_type(8)))  float  v8f;

__device__ __forceinline__ bf16 f2bf(float f) {
  union { float f; unsigned u; } in; in.f = f;
  unsigned short h =
      (unsigned short)((in.u + 0x7FFFu + ((in.u >> 16) & 1u)) >> 16);
  union { unsigned short s; bf16 b; } out; out.s = h;
  return out.b;
}

__device__ __forceinline__ float bf2f(bf16 b) {
  union { bf16 b; unsigned short s; } in; in.b = b;
  union { unsigned u; float f; } out; out.u = ((unsigned)in.s) << 16;
  return out.f;
}

__device__ __forceinline__ v16bf cat16(v8bf lo, v8bf hi) {
  return __builtin_shufflevector(lo, hi, 0, 1, 2, 3, 4, 5, 6, 7, 8, 9, 10, 11,
                                 12, 13, 14, 15);
}

// ---------------------------------------------------------------------------
// Stem: 7x7/2 conv (3->64) + BN + ReLU fused with 3x3/2 maxpool through LDS.
// Writes h0 in bf16 NHWC: h0[((img*56+y)*56+x)*64 + c].
// Grid: x = 16 pool tiles (4x4 of 14x14), y = 64 channels, z = 32 images.
// ---------------------------------------------------------------------------
__global__ void k_stem(const float* __restrict__ x, const float* __restrict__ w,
                       const float* __restrict__ bw, const float* __restrict__ bb,
                       const float* __restrict__ bm, const float* __restrict__ bv,
                       bf16* __restrict__ h0) {
  const int c = blockIdx.y, img = blockIdx.z;
  const int tile = blockIdx.x;
  const int ty0 = (tile >> 2) * 14, tx0 = (tile & 3) * 14;
  __shared__ float sw[147];      // 3*7*7 weights for this channel
  __shared__ float sc[29 * 29];  // conv outputs covering the pool tile
  const int t = threadIdx.x;
  if (t < 147) sw[t] = w[c * 147 + t];
  __syncthreads();
  const float inv  = bw[c] * rsqrtf(bv[c] + 1e-5f);
  const float bias = bb[c] - bm[c] * inv;
  const float* xi = x + (size_t)img * 3 * 224 * 224;
  for (int i = t; i < 29 * 29; i += 256) {
    const int cy = i / 29, cx = i % 29;
    const int gy = 2 * ty0 - 1 + cy;  // conv-grid coords (112x112)
    const int gx = 2 * tx0 - 1 + cx;
    float v = -__builtin_inff();
    if ((unsigned)gy < 112u && (unsigned)gx < 112u) {
      float s = 0.f;
      for (int ch = 0; ch < 3; ++ch)
        for (int ky = 0; ky < 7; ++ky) {
          const int iy = 2 * gy + ky - 3;
          if ((unsigned)iy >= 224u) continue;
          for (int kx = 0; kx < 7; ++kx) {
            const int ix = 2 * gx + kx - 3;
            if ((unsigned)ix < 224u)
              s += xi[(ch * 224 + iy) * 224 + ix] * sw[(ch * 7 + ky) * 7 + kx];
          }
        }
      v = fmaxf(s * inv + bias, 0.f);
    }
    sc[i] = v;
  }
  __syncthreads();
  if (t < 196) {
    const int py = t / 14, px = t % 14;
    float m = -__builtin_inff();
    for (int dy = 0; dy < 3; ++dy)
      for (int dx = 0; dx < 3; ++dx)
        m = fmaxf(m, sc[(2 * py + dy) * 29 + (2 * px + dx)]);
    h0[(((size_t)img * 56 + ty0 + py) * 56 + tx0 + px) * 64 + c] = f2bf(m);
  }
}

// ---------------------------------------------------------------------------
// Implicit-GEMM conv on WMMA bf16, bf16 NHWC activations.
// One wave32 per 32(Cout) x 32(pixel) output tile.
// Fragment element e (VGPR e/2, half e&1) of lane L (half = L>>4) holds
// k_local = (v<4 ? 2v : 2v+8) + (e&1) + 8*half, so elements 0..7 are the
// contiguous run {0..7}+8*half and 8..15 are {16..23}+8*half -> two b128
// loads per fragment for both A and B.
// ---------------------------------------------------------------------------
template <int KS>
__global__ void k_conv_wmma(const bf16* __restrict__ in, const bf16* __restrict__ wg,
                            const float* __restrict__ bni, const float* __restrict__ bnb,
                            const bf16* __restrict__ res, bf16* __restrict__ out,
                            int Cin, int Cout, int Hin, int Win, int Hout, int Wout,
                            int stride, int pad, int relu) {
  const int lane = threadIdx.x;
  const int half = lane >> 4, lidx = lane & 15;
  const int Ktot = Cin * KS * KS;
  const int mbase = blockIdx.y * 32;

  int p[2], yh[2], xw[2], img[2];
#pragma unroll
  for (int nf = 0; nf < 2; ++nf) {
    p[nf] = blockIdx.x * 32 + nf * 16 + lidx;
    xw[nf] = p[nf] % Wout;
    int t = p[nf] / Wout;
    yh[nf] = t % Hout;
    img[nf] = t / Hout;
  }
  const bf16* arow0 = wg + (size_t)(mbase + lidx) * Ktot + 8 * half;
  const bf16* arow1 = wg + (size_t)(mbase + 16 + lidx) * Ktot + 8 * half;
  const bf16 ZB = f2bf(0.f);

  v8f acc00 = {0.f,0.f,0.f,0.f,0.f,0.f,0.f,0.f};
  v8f acc01 = acc00, acc10 = acc00, acc11 = acc00;

#pragma unroll 1  // keep the tap loop rolled: bounded live set, no spills
  for (int kk = 0; kk < KS * KS; ++kk) {
    const int ky = (KS == 1) ? 0 : kk / KS;
    const int kx = (KS == 1) ? 0 : kk % KS;
    bool ok0, ok1;
    const bf16 *bp0, *bp1;
    {
      const int iy0 = yh[0] * stride + ky - pad;
      const int ix0 = xw[0] * stride + kx - pad;
      ok0 = (unsigned)iy0 < (unsigned)Hin && (unsigned)ix0 < (unsigned)Win;
      bp0 = in + (((size_t)img[0] * Hin + iy0) * Win + ix0) * Cin + 8 * half;
      const int iy1 = yh[1] * stride + ky - pad;
      const int ix1 = xw[1] * stride + kx - pad;
      ok1 = (unsigned)iy1 < (unsigned)Hin && (unsigned)ix1 < (unsigned)Win;
      bp1 = in + (((size_t)img[1] * Hin + iy1) * Win + ix1) * Cin + 8 * half;
    }
    const bf16* ap0 = arow0 + kk * Cin;
    const bf16* ap1 = arow1 + kk * Cin;
#pragma unroll 1  // rolled: 8 b128 loads + 4 wmma per iteration, ~80 live VGPRs
    for (int ci = 0; ci < Cin; ci += 32) {
      const v16bf a0 = cat16(*(const v8bf*)(ap0 + ci), *(const v8bf*)(ap0 + ci + 16));
      const v16bf a1 = cat16(*(const v8bf*)(ap1 + ci), *(const v8bf*)(ap1 + ci + 16));
      v16bf b0, b1;
      if (ok0) {
        b0 = cat16(*(const v8bf*)(bp0 + ci), *(const v8bf*)(bp0 + ci + 16));
      } else {
#pragma unroll
        for (int e = 0; e < 16; ++e) b0[e] = ZB;
      }
      if (ok1) {
        b1 = cat16(*(const v8bf*)(bp1 + ci), *(const v8bf*)(bp1 + ci + 16));
      } else {
#pragma unroll
        for (int e = 0; e < 16; ++e) b1[e] = ZB;
      }
      acc00 = __builtin_amdgcn_wmma_f32_16x16x32_bf16(false, a0, false, b0,
                                                      (short)0, acc00, false, false);
      acc01 = __builtin_amdgcn_wmma_f32_16x16x32_bf16(false, a0, false, b1,
                                                      (short)0, acc01, false, false);
      acc10 = __builtin_amdgcn_wmma_f32_16x16x32_bf16(false, a1, false, b0,
                                                      (short)0, acc10, false, false);
      acc11 = __builtin_amdgcn_wmma_f32_16x16x32_bf16(false, a1, false, b1,
                                                      (short)0, acc11, false, false);
    }
  }

  // Epilogue: folded BN, optional residual add, optional ReLU, one f32->bf16
  // per element, 16-byte vector stores (channels contiguous in NHWC).
#pragma unroll
  for (int mb = 0; mb < 2; ++mb) {
#pragma unroll
    for (int nf = 0; nf < 2; ++nf) {
      const v8f a = (mb == 0) ? (nf == 0 ? acc00 : acc01)
                              : (nf == 0 ? acc10 : acc11);
      const int cb = mbase + mb * 16 + 8 * half;
      const size_t ob = (size_t)p[nf] * Cout + cb;
      v8bf rv;
      if (res) rv = *(const v8bf*)(res + ob);
      v8bf ov;
#pragma unroll
      for (int r = 0; r < 8; ++r) {
        float vo = a[r] * bni[cb + r] + bnb[cb + r];
        if (res) vo += bf2f(rv[r]);
        if (relu) vo = fmaxf(vo, 0.f);
        ov[r] = f2bf(vo);
      }
      *(v8bf*)(out + ob) = ov;
    }
  }
}

// Gather sliced conv weights, f32 -> bf16, K order = (ky,kx) outer, ci inner:
// out[co*Ci*ks2 + kk*Ci + ci] = W[rows[co]][cols[ci]][kk]
__global__ void k_gather_w(const float* __restrict__ W, const int* __restrict__ rows,
                           const int* __restrict__ cols, bf16* __restrict__ out,
                           int Ci, int CinFull, int ks2, int tot) {
  const int i = blockIdx.x * 256 + threadIdx.x;
  if (i >= tot) return;
  const int ci = i % Ci;
  int t = i / Ci;
  const int kk = t % ks2;
  const int co = t / ks2;
  const int sc = cols ? cols[ci] : ci;
  out[i] = f2bf(W[((size_t)rows[co] * CinFull + sc) * ks2 + kk]);
}

// Gather + fold BN: inv = w/sqrt(v+eps), bias = b - m*inv (sliced channels).
__global__ void k_gather_bn(const float* __restrict__ w, const float* __restrict__ b,
                            const float* __restrict__ m, const float* __restrict__ v,
                            const int* __restrict__ idx, float* __restrict__ inv,
                            float* __restrict__ bias, int C) {
  const int i = blockIdx.x * 64 + threadIdx.x;
  if (i >= C) return;
  const int j = idx[i];
  const float iv = w[j] * rsqrtf(v[j] + 1e-5f);
  inv[i] = iv;
  bias[i] = b[j] - m[j] * iv;
}

// Stage-0 shortcut: gather channels of bf16 NHWC h0 into compact NHWC.
__global__ void k_gather_ch(const bf16* __restrict__ in, const int* __restrict__ idx,
                            bf16* __restrict__ out, int Csub, int CinFull,
                            long long tot) {
  const long long i = (long long)blockIdx.x * 256 + threadIdx.x;
  if (i >= tot) return;
  const int j = (int)(i % Csub);
  const long long pix = i / Csub;  // flattened n*HW+p
  out[i] = in[pix * CinFull + idx[j]];
}

// Head: 7x7 global mean per channel (bf16 NHWC), sliced FC, tanh, write if
// ctx_label == c (each image is covered by exactly one context).
__global__ void k_head(const bf16* __restrict__ feat, const float* __restrict__ fcw,
                       const float* __restrict__ fcb, const int* __restrict__ idx3c,
                       const int* __restrict__ ctx, int c, float* __restrict__ out) {
  const int n = blockIdx.x;
  const int j = threadIdx.x;  // 256 channels
  __shared__ float s[256];
  const bf16* p = feat + (size_t)n * 49 * 256 + j;
  float sum = 0.f;
  for (int q = 0; q < 49; ++q) sum += bf2f(p[q * 256]);
  s[j] = (sum * (1.0f / 49.0f)) * fcw[idx3c[j]];
  __syncthreads();
  for (int off = 128; off > 0; off >>= 1) {
    if (j < off) s[j] += s[j + off];
    __syncthreads();
  }
  if (j == 0 && ctx[n] == c) out[n] = tanhf(s[0] + fcb[0]);
}

// ---------------------------------------------------------------------------
// Host orchestration. Input leaves assumed flattened depth-first in dict
// insertion order of setup_inputs(): x; stem{conv_w, bn{w,b,m,v}}; stages
// (block dicts: conv1_w, bn1{w,b,m,v}, conv2_w, bn2{w,b,m,v}[, proj_w,
// proj_bn{w,b,m,v}]); fc_w; fc_b; ctx_label; idx0..idx3.  (108 leaves)
// ---------------------------------------------------------------------------
extern "C" void kernel_launch(void* const* d_in, const int* in_sizes, int n_in,
                              void* d_out, int out_size, void* d_ws, size_t ws_size,
                              hipStream_t stream) {
  (void)in_sizes; (void)n_in; (void)out_size; (void)ws_size;
  auto F = [&](int i) { return (const float*)d_in[i]; };
  auto I = [&](int i) { return (const int*)d_in[i]; };

  // -------- workspace carve (~41 MB) --------
  char* wp = (char*)d_ws;
  auto carve = [&](size_t bytes) {
    void* r = wp;
    wp += (bytes + 255) & ~(size_t)255;
    return r;
  };
  bf16* h0 = (bf16*)carve((size_t)32 * 64 * 56 * 56 * 2);  // 12.85 MB, NHWC bf16
  const size_t ABYTES = (size_t)32 * 32 * 56 * 56 * 2;     // 6.4 MB (max act)
  bf16* A0 = (bf16*)carve(ABYTES);
  bf16* A1 = (bf16*)carve(ABYTES);
  bf16* Tb = (bf16*)carve(ABYTES);
  bf16* SC = (bf16*)carve(ABYTES);
  bf16* wbuf = (bf16*)carve((size_t)256 * 256 * 9 * 2);    // 1.2 MB
  float* bni = (float*)carve(512 * 4);
  float* bnb = (float*)carve(512 * 4);

  // -------- stem + maxpool --------
  dim3 sg(16, 64, 32);
  k_stem<<<sg, 256, 0, stream>>>(F(0), F(1), F(2), F(3), F(4), F(5), h0);

  struct Blk { int c1, bn1, c2, bn2, pw, pbn; };
  static const Blk BL[4][2] = {
      {{6, 7, 11, 12, -1, -1},   {16, 17, 21, 22, -1, -1}},
      {{26, 27, 31, 32, 36, 37}, {41, 42, 46, 47, -1, -1}},
      {{51, 52, 56, 57, 61, 62}, {66, 67, 71, 72, -1, -1}},
      {{76, 77, 81, 82, 86, 87}, {91, 92, 96, 97, -1, -1}},
  };
  static const int KEEPc[4] = {32, 64, 128, 256};
  static const int FULLc[4] = {64, 128, 256, 512};
  static const int HS[4] = {56, 28, 14, 7};
  static const int IDXL[4] = {104, 105, 106, 107};
  const int L_FCW = 101, L_FCB = 102, L_CTX = 103;

  auto gw = [&](int leaf, const int* rows, const int* cols, int Co, int Ci,
                int CinFull, int ks2) {
    const int tot = Co * Ci * ks2;
    k_gather_w<<<(tot + 255) / 256, 256, 0, stream>>>(F(leaf), rows, cols, wbuf,
                                                      Ci, CinFull, ks2, tot);
  };
  auto gbn = [&](int leaf, const int* rows, int C) {
    k_gather_bn<<<(C + 63) / 64, 64, 0, stream>>>(F(leaf), F(leaf + 1),
                                                  F(leaf + 2), F(leaf + 3), rows,
                                                  bni, bnb, C);
  };
  auto conv = [&](const bf16* inp, const bf16* resp, bf16* outp, int Cin,
                  int Cout, int Hin, int Hout, int stride, int ks, int pad,
                  int relu) {
    dim3 g((32 * Hout * Hout) / 32, Cout / 32);
    if (ks == 3)
      k_conv_wmma<3><<<g, 32, 0, stream>>>(inp, wbuf, bni, bnb, resp, outp, Cin,
                                           Cout, Hin, Hin, Hout, Hout, stride,
                                           pad, relu);
    else
      k_conv_wmma<1><<<g, 32, 0, stream>>>(inp, wbuf, bni, bnb, resp, outp, Cin,
                                           Cout, Hin, Hin, Hout, Hout, stride,
                                           pad, relu);
  };

  for (int c = 0; c < 3; ++c) {
    const bf16* cur = h0;
    int curH = 56;
    const int* prevIdx = nullptr;  // stage-0 conv1 uses full 64 input channels
    int prevK = 64;
    for (int si = 0; si < 4; ++si) {
      const int k = KEEPc[si];
      const int* ci = I(IDXL[si]) + c * k;
      const int Ho = HS[si];
      const int stride = si ? 2 : 1;
      const Blk& b0 = BL[si][0];

      // shortcut for block 0
      if (si == 0) {
        const long long tot = (long long)32 * 56 * 56 * k;
        k_gather_ch<<<(unsigned)((tot + 255) / 256), 256, 0, stream>>>(
            h0, ci, SC, k, 64, tot);
      } else {
        gw(b0.pw, ci, prevIdx, k, prevK, FULLc[si - 1], 1);
        gbn(b0.pbn, ci, k);
        conv(cur, nullptr, SC, prevK, k, curH, Ho, 2, 1, 0, 0);
      }
      // block 0 conv1 (+bn1+relu)
      gw(b0.c1, ci, prevIdx, k, prevK, si == 0 ? 64 : FULLc[si - 1], 9);
      gbn(b0.bn1, ci, k);
      conv(cur, nullptr, Tb, prevK, k, curH, Ho, stride, 3, 1, 1);
      // block 0 conv2 (+bn2 + shortcut + relu)
      gw(b0.c2, ci, ci, k, k, FULLc[si], 9);
      gbn(b0.bn2, ci, k);
      conv(Tb, SC, A0, k, k, Ho, Ho, 1, 3, 1, 1);

      // block 1 (identity shortcut)
      const Blk& b1 = BL[si][1];
      gw(b1.c1, ci, ci, k, k, FULLc[si], 9);
      gbn(b1.bn1, ci, k);
      conv(A0, nullptr, Tb, k, k, Ho, Ho, 1, 3, 1, 1);
      gw(b1.c2, ci, ci, k, k, FULLc[si], 9);
      gbn(b1.bn2, ci, k);
      conv(Tb, A0, A1, k, k, Ho, Ho, 1, 3, 1, 1);

      cur = A1;
      curH = Ho;
      prevIdx = ci;
      prevK = k;
    }
    k_head<<<32, 256, 0, stream>>>(A1, F(L_FCW), F(L_FCB), I(107) + c * 256,
                                   I(L_CTX), c, (float*)d_out);
  }
}
